// TransformerLayerMOE_13546326852248
// MI455X (gfx1250) — matrix operand order repaired and checked by reading, not compiled, not run on hardware
//
#include <hip/hip_runtime.h>
#include <hip/hip_bf16.h>
#include <math.h>

// ---------------- problem constants (match reference) ----------------
#define TDIM   1024
#define THEADS 16
#define TDH    64
#define TFFN   4096
#define TE     8
#define TB     2
#define TS     1024
#define TT     (TB * TS)      // 2048 tokens

#define DEV __device__ __forceinline__

typedef __attribute__((ext_vector_type(16))) __bf16 v16bf;
typedef __attribute__((ext_vector_type(8)))  float  v8f;

union Frag16 { unsigned u[8]; v16bf v; };

DEV unsigned short f2bf(float f) {
  unsigned u = __float_as_uint(f);
  return (unsigned short)((u + 0x7FFFu + ((u >> 16) & 1u)) >> 16);
}
DEV float bf2f(unsigned short h) { return __uint_as_float(((unsigned)h) << 16); }
DEV unsigned pack2(float a, float b) {
  return (unsigned)f2bf(a) | ((unsigned)f2bf(b) << 16);
}
DEV float gelu_exact(float x) {
  return 0.5f * x * (1.0f + erff(x * 0.7071067811865475f));
}

// CDNA5 async copy global -> LDS (ASYNCcnt-tracked, no VGPR round trip)
DEV void async_g2l_b64(unsigned ldsAddr, const void* gaddr) {
  asm volatile("global_load_async_to_lds_b64 %0, %1, off"
               :: "v"(ldsAddr), "v"((unsigned long long)(uintptr_t)gaddr)
               : "memory");
}
DEV void wait_async0() {
  asm volatile("s_wait_asynccnt 0x0" ::: "memory");
}

// ---------------- generic WMMA bf16 GEMM ----------------
// C[z][M][N] = act(alpha * A[z][M][K] x B[z][K][N] + bias[z][N] (+ resid))
// Block tile 256(M) x 128(N), 8 waves, each wave 64x64 (16 WMMA / K-step).
// A: row-major (bf16 via async double-buffered LDS, or f32 converted in VGPRs).
// B modes: 0 = pre-swizzled weights [N/16][K/32][lane32][16 bf16]
//          1 = strided, K contiguous per row  (element = B[n*bSN + k])
//          2 = generic strided                (element = B[n*bSN + k*bSK])
struct GemmP {
  const void* A; long lda, aS1, aS2;
  const unsigned short* B; long bSN, bSK, bS1, bS2;
  const float* bias; long biasS;
  float* Cf; unsigned short* Cb; long ldc, cS1, cS2;
  const float* resid;
  int M, N, K; float alpha; int zdiv;
  const int* counts;            // per-z row counts for MoE early exit (or null)
};

#define LDS_ROW_U 18            // uints per padded A row (36 bf16)
#define LDS_BUF_U (256 * LDS_ROW_U)

template<bool AF32, int BMODE, bool DOGELU, bool RESID>
__global__ __launch_bounds__(256) void gemm_wmma(GemmP p) {
  const int z  = blockIdx.z;
  const int z1 = (p.zdiv > 1) ? (z / p.zdiv) : z;
  const int z2 = (p.zdiv > 1) ? (z % p.zdiv) : 0;
  const int mBase = blockIdx.y * 256;
  const int nBase = blockIdx.x * 128;
  if (p.counts) {
    const int c   = p.counts[z];
    const int lim = (c + 255) & ~255;
    if (mBase >= lim) return;
  }
  const size_t aoff = (size_t)z1 * p.aS1 + (size_t)z2 * p.aS2;
  const size_t boff = (size_t)z1 * p.bS1 + (size_t)z2 * p.bS2;
  const size_t coff = (size_t)z1 * p.cS1 + (size_t)z2 * p.cS2;

  // A tile: 256 rows x 32 cols bf16, padded row stride 36 elems; 2 buffers.
  __shared__ unsigned sA[2 * LDS_BUF_U];

  const int tid  = threadIdx.x;
  const int lane = tid & 31;
  const int wave = tid >> 5;
  const int wm   = wave & 3;     // 64-row slab within 256
  const int wn   = wave >> 2;    // 64-col slab within 128
  const int half = lane >> 4;
  const int l16  = lane & 15;

  v8f acc[4][4] = {};

  const int KT = p.K >> 5;
  const unsigned sAaddr = (unsigned)(uintptr_t)(&sA[0]);

  // async stage of one 256x32 bf16 tile: 2048 8-byte chunks, 8 per thread
  auto issueStage = [&](int kt, int buf) {
    const unsigned base = sAaddr + (unsigned)buf * (LDS_BUF_U * 4);
    const unsigned short* srcA =
        (const unsigned short*)p.A + aoff + (size_t)mBase * p.lda + (kt << 5);
    #pragma unroll
    for (int j = 0; j < 8; ++j) {
      const int c   = tid + (j << 8);
      const int row = c >> 3;
      const int c8  = c & 7;                 // 8B chunk within row
      async_g2l_b64(base + row * (LDS_ROW_U * 4) + (c8 << 3),
                    srcA + (size_t)row * p.lda + (c8 << 2));
    }
  };

  if (!AF32) {
    issueStage(0, 0);
    wait_async0();
    __syncthreads();
  }

  for (int kt = 0; kt < KT; ++kt) {
    const int cur = AF32 ? 0 : (kt & 1);
    if (AF32) {
      // manual stage with f32 -> bf16 conversion (one row per thread)
      if (kt) __syncthreads();
      const float* src = (const float*)p.A + aoff +
                         (size_t)(mBase + tid) * p.lda + (kt << 5);
      __builtin_prefetch(src + 32, 0, 0);
      unsigned* dst = &sA[tid * LDS_ROW_U];
      #pragma unroll
      for (int j = 0; j < 8; ++j) {
        float4 v = ((const float4*)src)[j];
        dst[2 * j]     = pack2(v.x, v.y);
        dst[2 * j + 1] = pack2(v.z, v.w);
      }
      __syncthreads();
    } else if (kt + 1 < KT) {
      issueStage(kt + 1, cur ^ 1);           // overlap DMA with WMMA below
    }

    const unsigned bufU = (unsigned)cur * LDS_BUF_U;
    const int k0 = kt << 5;

    // ---- A fragments (four 16-row tiles), contiguous k = 16*half + s layout ----
    Frag16 a[4];
    #pragma unroll
    for (int mt = 0; mt < 4; ++mt) {
      const int r = (wm << 6) + (mt << 4) + l16;
      #pragma unroll
      for (int i = 0; i < 8; ++i) {
        const int off = (half << 3) + i;
        a[mt].u[i] = sA[bufU + r * LDS_ROW_U + off];
      }
    }

    #pragma unroll
    for (int nt = 0; nt < 4; ++nt) {
      const int n = nBase + (wn << 6) + (nt << 4) + l16;
      Frag16 b;
      if (BMODE == 0) {
        const size_t tcol = (size_t)((nBase >> 4) + (wn << 2) + nt);
        const unsigned short* src =
            p.B + boff + ((tcol * KT + kt) * 32 + lane) * 16;
        uint4 v0 = ((const uint4*)src)[0];
        uint4 v1 = ((const uint4*)src)[1];
        b.u[0] = v0.x; b.u[1] = v0.y; b.u[2] = v0.z; b.u[3] = v0.w;
        b.u[4] = v1.x; b.u[5] = v1.y; b.u[6] = v1.z; b.u[7] = v1.w;
      } else if (BMODE == 1) {
        if (n < p.N) {
          const unsigned short* src =
              p.B + boff + (size_t)n * p.bSN + k0 + (half << 4);
          uint4 v0 = ((const uint4*)src)[0];
          uint4 v1 = ((const uint4*)src)[1];
          b.u[0] = v0.x; b.u[1] = v0.y; b.u[2] = v0.z; b.u[3] = v0.w;
          b.u[4] = v1.x; b.u[5] = v1.y; b.u[6] = v1.z; b.u[7] = v1.w;
        } else {
          #pragma unroll
          for (int i = 0; i < 8; ++i) b.u[i] = 0;
        }
      } else {
        if (n < p.N) {
          const size_t base = boff + (size_t)n * p.bSN +
                              (size_t)(k0 + (half << 4)) * p.bSK;
          #pragma unroll
          for (int i = 0; i < 8; ++i) {
            unsigned s0 = p.B[base + (size_t)(2 * i) * p.bSK];
            unsigned s1 = p.B[base + (size_t)(2 * i + 1) * p.bSK];
            b.u[i] = s0 | (s1 << 16);
          }
        } else {
          #pragma unroll
          for (int i = 0; i < 8; ++i) b.u[i] = 0;
        }
      }
      #pragma unroll
      for (int mt = 0; mt < 4; ++mt)
        acc[mt][nt] = __builtin_amdgcn_wmma_f32_16x16x32_bf16(
            false, a[mt].v, false, b.v, (short)0, acc[mt][nt], false, false);
    }

    if (!AF32) {
      wait_async0();          // next-buffer DMA complete (wave-local)
      __syncthreads();        // all waves' writes visible
    }
  }

  // ---- epilogue: alpha, bias, resid, GELU; C/D 16x16 f32 lane layout ----
  #pragma unroll
  for (int nt = 0; nt < 4; ++nt) {
    const int n = nBase + (wn << 6) + (nt << 4) + l16;
    if (n >= p.N) continue;
    const float bv = p.bias ? p.bias[(size_t)z * p.biasS + n] : 0.0f;
    #pragma unroll
    for (int mt = 0; mt < 4; ++mt) {
      #pragma unroll
      for (int j = 0; j < 8; ++j) {
        const int m = mBase + (wm << 6) + (mt << 4) + (half << 3) + j;
        float v = acc[mt][nt][j] * p.alpha + bv;
        const size_t ci = coff + (size_t)m * p.ldc + n;
        if (RESID) v += p.resid[ci];
        if (DOGELU) v = gelu_exact(v);
        if (p.Cf) p.Cf[ci] = v; else p.Cb[ci] = f2bf(v);
      }
    }
  }
}

// ---------------- weight fp32 -> bf16 WMMA-swizzle ----------------
// dst tile layout: [(nTile*KT + kTile)*32 + lane]*16 contiguous bf16
__global__ __launch_bounds__(256) void swizzle_w(const float* src,
                                                 unsigned short* dst,
                                                 int K, int N) {
  const size_t bofs = (size_t)blockIdx.z * K * N;
  src += bofs; dst += bofs;
  const int KT = K >> 5;
  const int gid = blockIdx.x * 256 + threadIdx.x;
  const int nTiles = KT * (N >> 4);
  const int tile = gid >> 5;
  if (tile >= nTiles) return;
  const int lane = gid & 31;
  const int kt = tile % KT;
  const int nt = tile / KT;
  const int kb = (kt << 5) + ((lane >> 4) << 4);
  const int n  = (nt << 4) + (lane & 15);
  unsigned short tmp[16];
  #pragma unroll
  for (int e = 0; e < 16; ++e) tmp[e] = f2bf(src[(size_t)(kb + e) * N + n]);
  unsigned* out = (unsigned*)(dst + ((size_t)tile * 32 + lane) * 16);
  #pragma unroll
  for (int e = 0; e < 16; e += 2)
    out[e >> 1] = (unsigned)tmp[e] | ((unsigned)tmp[e + 1] << 16);
}

// ---------------- LayerNorm -> bf16 ----------------
__global__ __launch_bounds__(256) void ln_k(const float* x, const float* g,
                                            const float* b, unsigned short* out,
                                            int d) {
  const int row = blockIdx.x;
  const float* xr = x + (size_t)row * d;
  __shared__ float red[256];
  float s = 0.f;
  for (int i = threadIdx.x; i < d; i += 256) s += xr[i];
  red[threadIdx.x] = s; __syncthreads();
  for (int o = 128; o > 0; o >>= 1) {
    if (threadIdx.x < o) red[threadIdx.x] += red[threadIdx.x + o];
    __syncthreads();
  }
  const float mu = red[0] / d;
  __syncthreads();
  float s2 = 0.f;
  for (int i = threadIdx.x; i < d; i += 256) { float t = xr[i] - mu; s2 += t * t; }
  red[threadIdx.x] = s2; __syncthreads();
  for (int o = 128; o > 0; o >>= 1) {
    if (threadIdx.x < o) red[threadIdx.x] += red[threadIdx.x + o];
    __syncthreads();
  }
  const float rstd = rsqrtf(red[0] / d + 1e-5f);
  unsigned short* orow = out + (size_t)row * d;
  for (int i = threadIdx.x; i < d; i += 256)
    orow[i] = f2bf((xr[i] - mu) * rstd * g[i] + b[i]);
}

// ---------------- softmax over f32 rows (in place) ----------------
__global__ __launch_bounds__(256) void softmax_k(float* s, int n) {
  float* r = s + (size_t)blockIdx.x * n;
  __shared__ float red[256];
  float mx = -3.4e38f;
  for (int i = threadIdx.x; i < n; i += 256) mx = fmaxf(mx, r[i]);
  red[threadIdx.x] = mx; __syncthreads();
  for (int o = 128; o > 0; o >>= 1) {
    if (threadIdx.x < o) red[threadIdx.x] = fmaxf(red[threadIdx.x], red[threadIdx.x + o]);
    __syncthreads();
  }
  mx = red[0]; __syncthreads();
  float sum = 0.f;
  for (int i = threadIdx.x; i < n; i += 256) {
    float e = __expf(r[i] - mx); r[i] = e; sum += e;
  }
  red[threadIdx.x] = sum; __syncthreads();
  for (int o = 128; o > 0; o >>= 1) {
    if (threadIdx.x < o) red[threadIdx.x] += red[threadIdx.x + o];
    __syncthreads();
  }
  const float inv = 1.0f / red[0];
  for (int i = threadIdx.x; i < n; i += 256) r[i] *= inv;
}

// ---------------- top-1 gate + mixing coefficients (one wave / token) ----
__global__ __launch_bounds__(256) void gate_k(const unsigned short* xn2,
                                              const float* gw, const float* cw,
                                              const float* cb, int* eidx,
                                              float* gval, float* coef,
                                              int d, int T) {
  const int wave = threadIdx.x >> 5;
  const int lane = threadIdx.x & 31;
  const int t = blockIdx.x * 8 + wave;
  if (t >= T) return;
  const unsigned short* xr = xn2 + (size_t)t * d;
  float a[TE] = {0, 0, 0, 0, 0, 0, 0, 0};
  float c0 = 0.f, c1 = 0.f;
  for (int i = lane; i < d; i += 32) {
    const float xv = bf2f(xr[i]);
    #pragma unroll
    for (int e = 0; e < TE; ++e) a[e] += xv * gw[(size_t)i * TE + e];
    c0 += xv * cw[(size_t)i * 2];
    c1 += xv * cw[(size_t)i * 2 + 1];
  }
  #pragma unroll
  for (int e = 0; e < TE; ++e)
    for (int o = 16; o; o >>= 1) a[e] += __shfl_xor(a[e], o, 32);
  for (int o = 16; o; o >>= 1) { c0 += __shfl_xor(c0, o, 32); c1 += __shfl_xor(c1, o, 32); }
  if (lane == 0) {
    float mx = a[0]; int bi = 0;
    #pragma unroll
    for (int e = 1; e < TE; ++e) if (a[e] > mx) { mx = a[e]; bi = e; }
    float sum = 0.f;
    #pragma unroll
    for (int e = 0; e < TE; ++e) sum += __expf(a[e] - mx);
    gval[t] = 1.0f / sum;     // max softmax prob
    eidx[t] = bi;
    c0 += cb[0]; c1 += cb[1];
    const float m2 = fmaxf(c0, c1);
    const float e0 = __expf(c0 - m2), e1 = __expf(c1 - m2);
    coef[2 * t]     = e0 / (e0 + e1);
    coef[2 * t + 1] = e1 / (e0 + e1);
  }
}

// ---------------- MoE dispatch helpers ----------------
__global__ void moe_zero_k(int* cursor) {
  if (threadIdx.x < TE) cursor[threadIdx.x] = 0;
}
__global__ __launch_bounds__(256) void moe_scatter_k(const int* eidx, int* cursor,
                                                     int* pos, int T, int cap) {
  const int t = blockIdx.x * 256 + threadIdx.x;
  if (t >= T) return;
  const int e = eidx[t];
  const int slot = atomicAdd(&cursor[e], 1);
  pos[t] = e * cap + slot;
}
__global__ __launch_bounds__(256) void moe_gather_k(const int* pos,
                                                    const unsigned short* xn2,
                                                    unsigned short* Xg, int d) {
  const int t = blockIdx.x;
  const unsigned* src = (const unsigned*)(xn2 + (size_t)t * d);
  unsigned* dst = (unsigned*)(Xg + (size_t)pos[t] * d);
  for (int i = threadIdx.x; i < (d >> 1); i += 256) dst[i] = src[i];
}
__global__ __launch_bounds__(256) void moe_pad_k(const int* cursor,
                                                 unsigned short* Xg,
                                                 int cap, int d) {
  const int e = blockIdx.x;
  const int c = cursor[e];
  int end = (c + 255) & ~255; if (end > cap) end = cap;   // 256-row blocks
  const size_t n = (size_t)(end - c) * d;
  unsigned* dst = (unsigned*)(Xg + ((size_t)e * cap + c) * d);
  for (size_t i = threadIdx.x; i < (n >> 1); i += 256) dst[i] = 0u;
}

// ---------------- final combine ----------------
__global__ __launch_bounds__(256) void combine_k(const float* x1, const float* Yg,
                                                 const float* rm, const int* pos,
                                                 const float* gval, const float* coef,
                                                 float* out, int d) {
  const int t = blockIdx.x;
  const float gc = gval[t] * coef[2 * t];
  const float c1 = coef[2 * t + 1];
  const size_t p = (size_t)pos[t] * d, b = (size_t)t * d;
  for (int i = threadIdx.x; i < d; i += 256)
    out[b + i] = x1[b + i] + Yg[p + i] * gc + rm[b + i] * c1;
}

// ---------------- host launcher ----------------
static inline void* wsAlloc(char*& w, size_t bytes) {
  void* p = (void*)w;
  w += (bytes + 255) & ~(size_t)255;
  return p;
}

extern "C" void kernel_launch(void* const* d_in, const int* in_sizes, int n_in,
                              void* d_out, int out_size, void* d_ws, size_t ws_size,
                              hipStream_t stream) {
  (void)in_sizes; (void)n_in; (void)out_size; (void)ws_size;
  const float* x      = (const float*)d_in[0];
  const float* ln1_g  = (const float*)d_in[1];
  const float* ln1_b  = (const float*)d_in[2];
  const float* Wqkv   = (const float*)d_in[3];
  const float* bqkv   = (const float*)d_in[4];
  const float* Wo     = (const float*)d_in[5];
  const float* bo     = (const float*)d_in[6];
  const float* ln2_g  = (const float*)d_in[7];
  const float* ln2_b  = (const float*)d_in[8];
  const float* gate_w = (const float*)d_in[9];
  const float* W1     = (const float*)d_in[10];
  const float* b1     = (const float*)d_in[11];
  const float* W2     = (const float*)d_in[12];
  const float* b2     = (const float*)d_in[13];
  const float* rW1    = (const float*)d_in[14];
  const float* rb1    = (const float*)d_in[15];
  const float* rW2    = (const float*)d_in[16];
  const float* rb2    = (const float*)d_in[17];
  const float* coef_w = (const float*)d_in[18];
  const float* coef_b = (const float*)d_in[19];
  float* out = (float*)d_out;

  char* w = (char*)d_ws;
  unsigned short* sWqkv = (unsigned short*)wsAlloc(w, (size_t)TDIM * 3 * TDIM * 2);
  unsigned short* sWo   = (unsigned short*)wsAlloc(w, (size_t)TDIM * TDIM * 2);
  unsigned short* sW1   = (unsigned short*)wsAlloc(w, (size_t)TE * TDIM * TFFN * 2);
  unsigned short* sW2   = (unsigned short*)wsAlloc(w, (size_t)TE * TFFN * TDIM * 2);
  unsigned short* srW1  = (unsigned short*)wsAlloc(w, (size_t)TDIM * TFFN * 2);
  unsigned short* srW2  = (unsigned short*)wsAlloc(w, (size_t)TFFN * TDIM * 2);
  unsigned short* xn    = (unsigned short*)wsAlloc(w, (size_t)TT * TDIM * 2);
  unsigned short* qkv   = (unsigned short*)wsAlloc(w, (size_t)TT * 3 * TDIM * 2);
  float*          scores= (float*)wsAlloc(w, (size_t)TB * THEADS * TS * TS * 4);
  unsigned short* ctx   = (unsigned short*)wsAlloc(w, (size_t)TT * TDIM * 2);
  float*          x1    = (float*)wsAlloc(w, (size_t)TT * TDIM * 4);
  unsigned short* xn2   = (unsigned short*)wsAlloc(w, (size_t)TT * TDIM * 2);
  unsigned short* Xg    = (unsigned short*)wsAlloc(w, (size_t)TE * TT * TDIM * 2);
  unsigned short* Hg    = (unsigned short*)wsAlloc(w, (size_t)TE * TT * TFFN * 2);
  float*          Yg    = (float*)wsAlloc(w, (size_t)TE * TT * TDIM * 4);
  unsigned short* rH    = (unsigned short*)wsAlloc(w, (size_t)TT * TFFN * 2);
  float*          rm    = (float*)wsAlloc(w, (size_t)TT * TDIM * 4);
  int*   eidx  = (int*)wsAlloc(w, TT * 4);
  float* gval  = (float*)wsAlloc(w, TT * 4);
  float* coef  = (float*)wsAlloc(w, 2 * TT * 4);
  int*   pos   = (int*)wsAlloc(w, TT * 4);
  int*   cursor= (int*)wsAlloc(w, TE * 4);

  auto swGrid = [](int K, int N) {
    return dim3((unsigned)(((K >> 5) * (N >> 4) * 32 + 255) / 256), 1, 1);
  };
  // --- weight conversion + swizzle ---
  swizzle_w<<<swGrid(TDIM, 3 * TDIM), 256, 0, stream>>>(Wqkv, sWqkv, TDIM, 3 * TDIM);
  swizzle_w<<<swGrid(TDIM, TDIM),     256, 0, stream>>>(Wo,   sWo,   TDIM, TDIM);
  { dim3 g = swGrid(TDIM, TFFN); g.z = TE;
    swizzle_w<<<g, 256, 0, stream>>>(W1, sW1, TDIM, TFFN); }
  { dim3 g = swGrid(TFFN, TDIM); g.z = TE;
    swizzle_w<<<g, 256, 0, stream>>>(W2, sW2, TFFN, TDIM); }
  swizzle_w<<<swGrid(TDIM, TFFN), 256, 0, stream>>>(rW1, srW1, TDIM, TFFN);
  swizzle_w<<<swGrid(TFFN, TDIM), 256, 0, stream>>>(rW2, srW2, TFFN, TDIM);

  // --- LN1 ---
  ln_k<<<TT, 256, 0, stream>>>(x, ln1_g, ln1_b, xn, TDIM);

  // --- QKV GEMM: [2048,1024] x [1024,3072] -> bf16 ---
  { GemmP p{}; p.A = xn; p.lda = TDIM;
    p.B = sWqkv; p.bias = bqkv;
    p.Cb = qkv; p.ldc = 3 * TDIM;
    p.M = TT; p.N = 3 * TDIM; p.K = TDIM; p.alpha = 1.f; p.zdiv = 1;
    gemm_wmma<false, 0, false, false><<<dim3(24, 8, 1), 256, 0, stream>>>(p); }

  // --- scores = (Q x K^T)/8, batched over 32 (b,h) ---
  { GemmP p{}; p.A = qkv; p.lda = 3 * TDIM;
    p.aS1 = (long)TS * 3 * TDIM; p.aS2 = TDH;
    p.B = qkv + TDIM; p.bSN = 3 * TDIM;
    p.bS1 = (long)TS * 3 * TDIM; p.bS2 = TDH;
    p.Cf = scores; p.ldc = TS;
    p.cS1 = (long)THEADS * TS * TS; p.cS2 = (long)TS * TS;
    p.M = TS; p.N = TS; p.K = TDH; p.alpha = 0.125f; p.zdiv = THEADS;
    gemm_wmma<false, 1, false, false><<<dim3(8, 4, TB * THEADS), 256, 0, stream>>>(p); }

  // --- softmax rows ---
  softmax_k<<<TB * THEADS * TS, 256, 0, stream>>>(scores, TS);

  // --- ctx = P x V ---
  { GemmP p{}; p.A = scores; p.lda = TS;
    p.aS1 = (long)THEADS * TS * TS; p.aS2 = (long)TS * TS;
    p.B = qkv + 2 * TDIM; p.bSN = 1; p.bSK = 3 * TDIM;
    p.bS1 = (long)TS * 3 * TDIM; p.bS2 = TDH;
    p.Cb = ctx; p.ldc = TDIM;
    p.cS1 = (long)TS * TDIM; p.cS2 = TDH;
    p.M = TS; p.N = TDH; p.K = TS; p.alpha = 1.f; p.zdiv = THEADS;
    gemm_wmma<true, 2, false, false><<<dim3(1, 4, TB * THEADS), 256, 0, stream>>>(p); }

  // --- x1 = x + ctx x Wo + bo ---
  { GemmP p{}; p.A = ctx; p.lda = TDIM;
    p.B = sWo; p.bias = bo;
    p.Cf = x1; p.ldc = TDIM; p.resid = x;
    p.M = TT; p.N = TDIM; p.K = TDIM; p.alpha = 1.f; p.zdiv = 1;
    gemm_wmma<false, 0, false, true><<<dim3(8, 8, 1), 256, 0, stream>>>(p); }

  // --- LN2 + gating + dispatch ---
  ln_k<<<TT, 256, 0, stream>>>(x1, ln2_g, ln2_b, xn2, TDIM);
  gate_k<<<TT / 8, 256, 0, stream>>>(xn2, gate_w, coef_w, coef_b,
                                     eidx, gval, coef, TDIM, TT);
  moe_zero_k<<<1, 32, 0, stream>>>(cursor);
  moe_scatter_k<<<TT / 256, 256, 0, stream>>>(eidx, cursor, pos, TT, TT);
  moe_gather_k<<<TT, 256, 0, stream>>>(pos, xn2, Xg, TDIM);
  moe_pad_k<<<TE, 256, 0, stream>>>(cursor, Xg, TT, TDIM);

  // --- MoE expert GEMM1 (GELU) ---
  { GemmP p{}; p.A = Xg; p.lda = TDIM; p.aS1 = (long)TT * TDIM;
    p.B = sW1; p.bS1 = (long)TDIM * TFFN;
    p.bias = b1; p.biasS = TFFN;
    p.Cb = Hg; p.ldc = TFFN; p.cS1 = (long)TT * TFFN;
    p.M = TT; p.N = TFFN; p.K = TDIM; p.alpha = 1.f; p.zdiv = 1;
    p.counts = cursor;
    gemm_wmma<false, 0, true, false><<<dim3(32, 8, TE), 256, 0, stream>>>(p); }

  // --- MoE expert GEMM2 ---
  { GemmP p{}; p.A = Hg; p.lda = TFFN; p.aS1 = (long)TT * TFFN;
    p.B = sW2; p.bS1 = (long)TFFN * TDIM;
    p.bias = b2; p.biasS = TDIM;
    p.Cf = Yg; p.ldc = TDIM; p.cS1 = (long)TT * TDIM;
    p.M = TT; p.N = TDIM; p.K = TFFN; p.alpha = 1.f; p.zdiv = 1;
    p.counts = cursor;
    gemm_wmma<false, 0, false, false><<<dim3(8, 8, TE), 256, 0, stream>>>(p); }

  // --- residual dense MLP ---
  { GemmP p{}; p.A = xn2; p.lda = TDIM;
    p.B = srW1; p.bias = rb1;
    p.Cb = rH; p.ldc = TFFN;
    p.M = TT; p.N = TFFN; p.K = TDIM; p.alpha = 1.f; p.zdiv = 1;
    gemm_wmma<false, 0, true, false><<<dim3(32, 8, 1), 256, 0, stream>>>(p); }
  { GemmP p{}; p.A = rH; p.lda = TFFN;
    p.B = srW2; p.bias = rb2;
    p.Cf = rm; p.ldc = TDIM;
    p.M = TT; p.N = TDIM; p.K = TFFN; p.alpha = 1.f; p.zdiv = 1;
    gemm_wmma<false, 0, false, false><<<dim3(8, 8, 1), 256, 0, stream>>>(p); }

  // --- combine ---
  combine_k<<<TT, 256, 0, stream>>>(x1, Yg, rm, pos, gval, coef, out, TDIM);
}
